// GodModeCausalGRU_11854109737084
// MI455X (gfx1250) — matrix operand order repaired and checked by reading, not compiled, not run on hardware
//
#include <hip/hip_runtime.h>
#include <math.h>

// ---------------------------------------------------------------------------
// GodModeCausalGRU for MI455X (gfx1250, wave32, WMMA).
// All GEMMs + per-step GRU hidden matvec use v_wmma_f32_16x16x32_bf16 with
// f32 accumulate. Recurrence = persistent 48-block cooperative kernel/layer.
// A-tile staging uses GLOBAL_LOAD_ASYNC_TO_LDS_B128 (+ s_wait_asynccnt).
// ---------------------------------------------------------------------------

#define DM   512
#define TT   1024
#define BBATCH 32

typedef __bf16 bf16;
typedef __attribute__((ext_vector_type(16))) __bf16 v16bf;
typedef __attribute__((ext_vector_type(8)))  __bf16 v8bf;
typedef __attribute__((ext_vector_type(2)))  __bf16 v2bf;
typedef __attribute__((ext_vector_type(8)))  float  v8f;
typedef __attribute__((ext_vector_type(4)))  float  v4f;
typedef __attribute__((ext_vector_type(4)))  int    v4i;

union V16 { v16bf v; v8bf h[2]; v2bf p[8]; };

#if defined(__gfx1250__) && __has_builtin(__builtin_amdgcn_global_load_async_to_lds_b128)
#define HAVE_ASYNC_LDS 1
typedef __attribute__((address_space(1))) v4i* gvec4p;
typedef __attribute__((address_space(3))) v4i* lvec4p;
#else
#define HAVE_ASYNC_LDS 0
#endif

// Let the compiler lower f32->bf16 (RNE); picks a native cvt op if available.
__device__ __forceinline__ bf16 f2bf(float f) { return (bf16)f; }

__device__ __forceinline__ v2bf pk2(float lo, float hi) {
#if __has_builtin(__builtin_amdgcn_cvt_pk_bf16_f32)
  return __builtin_amdgcn_cvt_pk_bf16_f32(lo, hi);
#else
  v2bf r; r[0] = (bf16)lo; r[1] = (bf16)hi; return r;
#endif
}

// pack 16 f32 (four v4f) -> v16bf
__device__ __forceinline__ v16bf pack16(v4f a0, v4f a1, v4f a2, v4f a3) {
  V16 u;
  u.p[0] = pk2(a0[0], a0[1]); u.p[1] = pk2(a0[2], a0[3]);
  u.p[2] = pk2(a1[0], a1[1]); u.p[3] = pk2(a1[2], a1[3]);
  u.p[4] = pk2(a2[0], a2[1]); u.p[5] = pk2(a2[2], a2[3]);
  u.p[6] = pk2(a3[0], a3[1]); u.p[7] = pk2(a3[2], a3[3]);
  return u.v;
}

__device__ __forceinline__ v8f wmma_bf16(v16bf a, v16bf b, v8f c) {
  return __builtin_amdgcn_wmma_f32_16x16x32_bf16(false, a, false, b, (short)0, c,
                                                 false, false);
}

__device__ __forceinline__ float geluf(float x) {
  return 0.5f * x * (1.0f + erff(x * 0.70710678118654752f));
}
__device__ __forceinline__ float sigm(float x) { return 1.0f / (1.0f + __expf(-x)); }

// Grid-wide barrier (monotonic counter; all blocks co-resident).
__device__ __forceinline__ void grid_barrier(int* cnt, int nb) {
  __syncthreads();
  if (threadIdx.x == 0) {
    __threadfence();
    int old = atomicAdd(cnt, 1);
    int target = (old / nb + 1) * nb;
    while (__hip_atomic_load(cnt, __ATOMIC_ACQUIRE, __HIP_MEMORY_SCOPE_AGENT) < target) {
      __builtin_amdgcn_s_sleep(2);
    }
  }
  __syncthreads();
}

// ---------------------------------------------------------------------------
// WMMA GEMM: C[M x N] (+)= A[M x K] * W^T + bias, W = N rows x K.
//  A(m,k) at A[(m+ashift)*lda + k]; rows with (m%period)+ashift<0 read as 0.
//  W(n,k) at W[n*wrow + k*wstride]  (wstride=7 = causal-conv tap view).
// Block: 256 threads (8 waves) -> 32(M) x 128(N); each wave owns one A frag
// and two N-tiles (two back-to-back v_wmma per K step). A chunk staged in LDS
// as f32 via async-load-to-LDS.
// ---------------------------------------------------------------------------
__global__ __launch_bounds__(256)
void gemm_kernel(const float* __restrict__ A, int lda, int ashift, int period,
                 const float* __restrict__ W, int wrow, int wstride,
                 const float* __restrict__ bias,
                 float* __restrict__ C, int ldc, int accum,
                 int M, int N, int K)
{
  __shared__ float Asf[32][36];               // 32 rows x 32 k (f32, +pad)
  const int tid = threadIdx.x;
  const int m0 = blockIdx.x * 32;
  const int n0 = blockIdx.y * 128;
  const int wave = tid >> 5, lane = tid & 31;
  const int lane16 = lane & 15, half = lane >> 4;
  const int m_tl = wave & 1, n_tl = wave >> 1;      // 2(M) x 4(N-pairs)
  v8f acc0 = {0.f,0.f,0.f,0.f,0.f,0.f,0.f,0.f};
  v8f acc1 = {0.f,0.f,0.f,0.f,0.f,0.f,0.f,0.f};

  for (int kc = 0; kc < K; kc += 32) {
    { // cooperative A staging: 16B per thread
      int e0 = tid * 4;
      int r = e0 >> 5, c = e0 & 31;
      int grow = m0 + r;
      int pos = (grow % period) + ashift;
      if (pos >= 0 && pos < period) {
        const float* ap = A + (size_t)(grow + ashift) * lda + kc + c;
#if HAVE_ASYNC_LDS
        __builtin_amdgcn_global_load_async_to_lds_b128(
            (gvec4p)(ap), (lvec4p)(&Asf[r][c]), 0, 0);
#else
        *(v4f*)&Asf[r][c] = *(const v4f*)ap;
#endif
        if (kc + 32 < K) __builtin_prefetch(ap + 32, 0, 1);   // global_prefetch
      } else {
        v4f z = {0.f, 0.f, 0.f, 0.f};
        *(v4f*)&Asf[r][c] = z;
      }
#if HAVE_ASYNC_LDS && __has_builtin(__builtin_amdgcn_s_wait_asynccnt)
      __builtin_amdgcn_s_wait_asynccnt(0);
#endif
    }
    __syncthreads();

    // B fragments: two 16-col tiles per wave, one W-row K-chunk per lane.
    V16 bb0, bb1;
    {
      int wr0 = n0 + n_tl * 32 + lane16;
      int wr1 = wr0 + 16;
      const float* wp0 = W + (size_t)wr0 * wrow + (size_t)(kc + half * 16) * wstride;
      const float* wp1 = W + (size_t)wr1 * wrow + (size_t)(kc + half * 16) * wstride;
      if (wstride == 1) {
        bb0.v = pack16(*(const v4f*)(wp0), *(const v4f*)(wp0 + 4),
                       *(const v4f*)(wp0 + 8), *(const v4f*)(wp0 + 12));
        bb1.v = pack16(*(const v4f*)(wp1), *(const v4f*)(wp1 + 4),
                       *(const v4f*)(wp1 + 8), *(const v4f*)(wp1 + 12));
      } else {
        for (int i = 0; i < 8; ++i) {
          bb0.p[i] = pk2(wp0[(size_t)(2*i) * wstride], wp0[(size_t)(2*i+1) * wstride]);
          bb1.p[i] = pk2(wp1[(size_t)(2*i) * wstride], wp1[(size_t)(2*i+1) * wstride]);
        }
      }
    }
    // A fragment from LDS (f32 -> bf16)
    V16 aa;
    {
      int ar = m_tl * 16 + lane16;
      int off = half * 16;
      aa.v = pack16(*(const v4f*)&Asf[ar][off],      *(const v4f*)&Asf[ar][off + 4],
                    *(const v4f*)&Asf[ar][off + 8], *(const v4f*)&Asf[ar][off + 12]);
    }
    acc0 = wmma_bf16(aa.v, bb0.v, acc0);
    acc1 = wmma_bf16(aa.v, bb1.v, acc1);
    __syncthreads();
  }

  // epilogue (C layout: vgpr v, lanes0-15 -> M=v, lanes16-31 -> M=v+8)
  int c0 = n0 + n_tl * 32 + lane16;
  int c1 = c0 + 16;
  float bv0 = bias ? bias[c0] : 0.0f;
  float bv1 = bias ? bias[c1] : 0.0f;
  for (int v = 0; v < 8; ++v) {
    int row = m0 + m_tl * 16 + v + half * 8;
    if (row < M) {
      size_t i0 = (size_t)row * ldc + c0;
      size_t i1 = (size_t)row * ldc + c1;
      float x0 = acc0[v] + bv0, x1 = acc1[v] + bv1;
      if (accum) { x0 += C[i0]; x1 += C[i1]; }
      C[i0] = x0; C[i1] = x1;
    }
  }
}

// ---------------------------------------------------------------------------
// Persistent LayerNorm-GRU scan. grid = 48 blocks x 128 threads (co-resident).
// Block i owns w_hh rows [32i,32i+32): blocks 0-15 r, 16-31 z, 32-47 n.
// ---------------------------------------------------------------------------
__global__ __launch_bounds__(128)
void gru_scan_kernel(const float* __restrict__ xg,
                     const float* __restrict__ whh, const float* __restrict__ bhh,
                     const float* __restrict__ lnrw, const float* __restrict__ lnrb,
                     const float* __restrict__ lnzw, const float* __restrict__ lnzb,
                     const float* __restrict__ lnnw, const float* __restrict__ lnnb,
                     float* __restrict__ h, float* __restrict__ seq,
                     float* __restrict__ S, float* __restrict__ stats,
                     int* __restrict__ sync, int T)
{
  __shared__ bf16  Wl[32][520];          // 32 w_hh rows x 512 k (+pad) bf16
  __shared__ float ubuf[1024], zbuf[1024];
  __shared__ float bsum[32], bsq[32];

  const int tid = threadIdx.x, blk = blockIdx.x, nb = gridDim.x;
  const int wave = tid >> 5, lane = tid & 31;
  const int lane16 = lane & 15, half = lane >> 4;
  const int m_tl = wave & 1, n_tl = wave >> 1;     // 4 waves -> 2(M) x 2(N)
  const int gate = blk >> 4;                       // 0=r 1=z 2=n
  const int gbase = blk * 32;
  const float eps = 1e-5f, invd = 1.0f / 512.0f;

  for (int i = tid; i < 32 * 512; i += 128) {      // preload weight slice
    int r = i >> 9, c = i & 511;
    Wl[r][c] = f2bf(whh[(size_t)(gbase + r) * 512 + c]);
  }
  __syncthreads();

  for (int t = 0; t < T; ++t) {
    // ---------------- phase A: hg = h @ whh_slice^T ----------------
    v8f acc = {0.f,0.f,0.f,0.f,0.f,0.f,0.f,0.f};
#pragma unroll 2
    for (int k0 = 0; k0 < 512; k0 += 32) {
      V16 aa, bb;
      int brow = m_tl * 16 + lane16;
      const float* hp = h + (size_t)brow * 512 + k0;
      int off = half * 16;
      aa.v = pack16(*(const v4f*)(hp + off),      *(const v4f*)(hp + off + 4),
                    *(const v4f*)(hp + off + 8), *(const v4f*)(hp + off + 12));
      int wr = n_tl * 16 + lane16;
      int kb = k0 + half * 16;
      bb.h[0] = *(const v8bf*)&Wl[wr][kb];
      bb.h[1] = *(const v8bf*)&Wl[wr][kb + 8];
      acc = wmma_bf16(aa.v, bb.v, acc);
    }
    if (tid < 32) { bsum[tid] = 0.f; bsq[tid] = 0.f; }
    __syncthreads();

    int colw = gbase + n_tl * 16 + lane16;         // 0..1535
    float bh = bhh[colw];
    for (int v = 0; v < 8; ++v) {
      int b = m_tl * 16 + v + half * 8;            // batch row 0..31
      float sv = acc[v] + bh;
      if (gate < 2) sv += xg[((size_t)b * T + t) * 1536 + colw];
      S[(size_t)b * 1536 + colw] = sv;
      if (gate < 2) {
        float s1 = sv, s2 = sv * sv;
        for (int o = 8; o; o >>= 1) { s1 += __shfl_xor(s1, o, 16); s2 += __shfl_xor(s2, o, 16); }
        if (lane16 == 0) { atomicAdd(&bsum[b], s1); atomicAdd(&bsq[b], s2); }  // ds_add_f32
      }
    }
    __syncthreads();
    if (gate < 2 && tid < 32) {
      float* st = stats + (size_t)(t * 3 + gate) * 64;
      atomicAdd(&st[tid], bsum[tid]);
      atomicAdd(&st[32 + tid], bsq[tid]);
    }
    grid_barrier(sync, nb);

    // ---------------- phase B: r, z, u = xn + r*hn ----------------
    if (blk < 16) {
      if (tid < 32) { bsum[tid] = 0.f; bsq[tid] = 0.f; }
      __syncthreads();
      const float* str = stats + (size_t)(t * 3 + 0) * 64;
      const float* stz = stats + (size_t)(t * 3 + 1) * 64;
      for (int it = 0; it < 8; ++it) {
        int idx = tid + it * 128;
        int b = idx >> 5, jl = idx & 31;
        int j = gbase + jl;                        // 0..511
        float mr = str[b] * invd, vr = str[32 + b] * invd - mr * mr;
        float sr = S[(size_t)b * 1536 + j];
        float r = sigm((sr - mr) * rsqrtf(vr + eps) * lnrw[j] + lnrb[j]);
        float mz = stz[b] * invd, vz = stz[32 + b] * invd - mz * mz;
        float sz = S[(size_t)b * 1536 + 512 + j];
        float z = sigm((sz - mz) * rsqrtf(vz + eps) * lnzw[j] + lnzb[j]);
        float hn = S[(size_t)b * 1536 + 1024 + j];
        float xn = xg[((size_t)b * T + t) * 1536 + 1024 + j];
        float u = xn + r * hn;
        ubuf[idx] = u; zbuf[idx] = z;
        atomicAdd(&bsum[b], u); atomicAdd(&bsq[b], u * u);
      }
      __syncthreads();
      if (tid < 32) {
        float* st = stats + (size_t)(t * 3 + 2) * 64;
        atomicAdd(&st[tid], bsum[tid]);
        atomicAdd(&st[32 + tid], bsq[tid]);
      }
    }
    grid_barrier(sync, nb);

    // ---------------- phase C: n, h update ----------------
    if (blk < 16) {
      const float* stn = stats + (size_t)(t * 3 + 2) * 64;
      for (int it = 0; it < 8; ++it) {
        int idx = tid + it * 128;
        int b = idx >> 5, jl = idx & 31;
        int j = gbase + jl;
        float mn = stn[b] * invd, vn = stn[32 + b] * invd - mn * mn;
        float n = tanhf((ubuf[idx] - mn) * rsqrtf(vn + eps) * lnnw[j] + lnnb[j]);
        float z = zbuf[idx];
        float hv = h[(size_t)b * 512 + j];
        float hnew = (1.0f - z) * n + z * hv;
        h[(size_t)b * 512 + j] = hnew;
        seq[((size_t)b * T + t) * 512 + j] = hnew;
      }
    }
    grid_barrier(sync, nb);
  }
}

// ----------------------------- small kernels -------------------------------

__global__ void zero_kernel(float* p, size_t n) {
  size_t i = (size_t)blockIdx.x * blockDim.x + threadIdx.x;
  size_t st = (size_t)gridDim.x * blockDim.x;
  for (; i < n; i += st) p[i] = 0.0f;
}

__global__ void embed_kernel(const float* __restrict__ flux, const float* __restrict__ dt,
                             const float* __restrict__ fpw, const float* __restrict__ fpb,
                             float* __restrict__ X, int total) {
  int i = blockIdx.x * blockDim.x + threadIdx.x;
  if (i >= total) return;
  int row = i >> 9, col = i & 511;
  float v;
  if (col < 256) {
    v = flux[row] * fpw[col] + fpb[col];
  } else {
    int c2 = col - 256;
    int k = c2 & 127;
    float d = fabsf(dt[row]) + 1e-6f;
    float div = __expf(-(float)(2 * k) * 0.07195578435503901f);  // ln(1e4)/128
    float a = log1pf(d) * div;
    v = (c2 < 128) ? __sinf(a) : __cosf(a);
  }
  X[i] = v;
}

__global__ __launch_bounds__(256)
void ln_act_kernel(const float* __restrict__ in, int istride,
                   float* __restrict__ out, int ostride,
                   const float* __restrict__ w, const float* __restrict__ b,
                   int d, int act) {
  int row = blockIdx.x, tid = threadIdx.x;
  __shared__ float rs[256], rq[256];
  const float* ip = in + (size_t)row * istride;
  float s = 0.f, q = 0.f;
  for (int c = tid; c < d; c += 256) { float x = ip[c]; s += x; q += x * x; }
  rs[tid] = s; rq[tid] = q; __syncthreads();
  for (int o = 128; o; o >>= 1) {
    if (tid < o) { rs[tid] += rs[tid + o]; rq[tid] += rq[tid + o]; }
    __syncthreads();
  }
  float mean = rs[0] / d;
  float inv = rsqrtf(rq[0] / d - mean * mean + 1e-5f);
  float* op = out + (size_t)row * ostride;
  for (int c = tid; c < d; c += 256) {
    float x = (ip[c] - mean) * inv * w[c] + b[c];
    if (act == 1) x = geluf(x);
    op[c] = x;
  }
}

__global__ void glu_kernel(const float* __restrict__ hb, float* __restrict__ out, int total) {
  int i = blockIdx.x * blockDim.x + threadIdx.x;
  if (i >= total) return;
  int row = i >> 9, col = i & 511;
  float a = hb[(size_t)row * 1024 + col];
  float g = hb[(size_t)row * 1024 + 512 + col];
  out[i] = a * sigm(g);
}

__global__ void tanh_kernel(float* x, int n) {
  int i = blockIdx.x * blockDim.x + threadIdx.x;
  if (i < n) x[i] = tanhf(x[i]);
}

__global__ void add_kernel(float* y, const float* x, int n) {
  int i = blockIdx.x * blockDim.x + threadIdx.x;
  if (i < n) y[i] += x[i];
}

__global__ __launch_bounds__(256)
void score_kernel(const float* __restrict__ s1, const float* __restrict__ w2,
                  const float* __restrict__ b2, float* __restrict__ sc) {
  int row = blockIdx.x, tid = threadIdx.x;
  __shared__ float rs[256];
  rs[tid] = s1[(size_t)row * 256 + tid] * w2[tid];
  __syncthreads();
  for (int o = 128; o; o >>= 1) { if (tid < o) rs[tid] += rs[tid + o]; __syncthreads(); }
  if (tid == 0) sc[row] = rs[0] + b2[0];
}

__global__ __launch_bounds__(256)
void softmax_pool_kernel(const float* __restrict__ sc, const float* __restrict__ X,
                         float* __restrict__ pooled, int T) {
  int b = blockIdx.x, tid = threadIdx.x;
  __shared__ float w[1024];
  __shared__ float red[256];
  const float* srow = sc + (size_t)b * T;
  float mx = -1e30f;
  for (int t = tid; t < T; t += 256) mx = fmaxf(mx, srow[t]);
  red[tid] = mx; __syncthreads();
  for (int o = 128; o; o >>= 1) { if (tid < o) red[tid] = fmaxf(red[tid], red[tid + o]); __syncthreads(); }
  mx = red[0]; __syncthreads();
  float s = 0.f;
  for (int t = tid; t < T; t += 256) { float e = __expf(srow[t] - mx); w[t] = e; s += e; }
  red[tid] = s; __syncthreads();
  for (int o = 128; o; o >>= 1) { if (tid < o) red[tid] += red[tid + o]; __syncthreads(); }
  float inv = 1.0f / (red[0] * (1.0f + 1e-8f));
  __syncthreads();
  for (int t = tid; t < T; t += 256) w[t] *= inv;
  __syncthreads();
  for (int d = tid; d < 512; d += 256) {
    float a = 0.f;
    for (int t = 0; t < T; ++t) a += w[t] * X[((size_t)b * T + t) * 512 + d];
    pooled[(size_t)b * 512 + d] = a;
  }
}

__global__ void cls_kernel(const float* __restrict__ pooled, const float* __restrict__ w,
                           const float* __restrict__ b, float* __restrict__ out) {
  int t = threadIdx.x;
  if (t < 96) {
    int bb = t / 3, c = t % 3;
    float s = 0.f;
    for (int k = 0; k < 512; ++k) s += pooled[(size_t)bb * 512 + k] * w[(size_t)c * 512 + k];
    out[t] = s + b[c];
  }
}

// ---------------------------------------------------------------------------

extern "C" void kernel_launch(void* const* d_in, const int* in_sizes, int n_in,
                              void* d_out, int out_size, void* d_ws, size_t ws_size,
                              hipStream_t stream) {
  (void)in_sizes; (void)n_in; (void)out_size; (void)ws_size;

  const float* flux = (const float*)d_in[0];
  const float* dtt  = (const float*)d_in[1];
  // d_in[2] = mask (all true) -> ignored
  const float* fpw   = (const float*)d_in[3];
  const float* fpb   = (const float*)d_in[4];
  const float* mixw  = (const float*)d_in[5];
  const float* mixb  = (const float*)d_in[6];
  const float* mixlnw= (const float*)d_in[7];
  const float* mixlnb= (const float*)d_in[8];
  const float* few1  = (const float*)d_in[9];
  const float* feb1  = (const float*)d_in[10];
  const float* feln1w= (const float*)d_in[11];
  const float* feln1b= (const float*)d_in[12];
  const float* few2  = (const float*)d_in[13];
  const float* feb2  = (const float*)d_in[14];
  const float* feln2w= (const float*)d_in[15];
  const float* feln2b= (const float*)d_in[16];
  const float* winw  = (const float*)d_in[17];
  const float* winb  = (const float*)d_in[18];
  const float* winlnw= (const float*)d_in[19];
  const float* winlnb= (const float*)d_in[20];
  const float* resp  = (const float*)d_in[21];
  const float* attw1 = (const float*)d_in[22];
  const float* attb1 = (const float*)d_in[23];
  const float* attw2 = (const float*)d_in[24];
  const float* attb2 = (const float*)d_in[25];
  const float* clsw  = (const float*)d_in[26];
  const float* clsb  = (const float*)d_in[27];
  const float* g[3][10];
  for (int L = 0; L < 3; ++L)
    for (int k = 0; k < 10; ++k) g[L][k] = (const float*)d_in[28 + L * 10 + k];
  // per-layer: w_ih, b_ih, w_hh, b_hh, ln_r_w, ln_r_b, ln_z_w, ln_z_b, ln_n_w, ln_n_b

  const size_t R = (size_t)BBATCH * TT;  // 32768 rows
  float* B0 = (float*)d_ws;
  float* B1 = B0 + R * 512;
  float* B2 = B1 + R * 512;              // rnn_in (feat | win), stride 1024
  float* B3 = B2 + R * 1024;             // xg, stride 1536
  float* B4 = B3 + R * 1536;
  float* B5 = B4 + R * 512;
  float* S  = B5 + R * 512;              // 32 x 1536 hg staging
  float* stats = S + 32 * 1536;
  const size_t statsL = (size_t)TT * 3 * 64;
  float* h0 = stats + 3 * statsL;
  float* h1 = h0 + 32 * 512;
  float* h2 = h1 + 32 * 512;
  float* pooled = h2 + 32 * 512;
  float* sbuf = pooled + 32 * 512;
  int* sync = (int*)(sbuf + R);

  size_t zn = (size_t)((sbuf + R + 64) - stats);
  zero_kernel<<<1024, 256, 0, stream>>>(stats, zn);

  auto gemm = [&](const float* A, int lda, int ashift,
                  const float* W, int wrow, int wstride,
                  const float* bias, float* C, int ldc, int accum, int N, int K) {
    dim3 grid((unsigned)(R / 32), (unsigned)(N / 128));
    gemm_kernel<<<grid, 256, 0, stream>>>(A, lda, ashift, TT, W, wrow, wstride,
                                          bias, C, ldc, accum, (int)R, N, K);
  };
  const int total = (int)(R * 512);
  const int eb = (total + 255) / 256;

  embed_kernel<<<eb, 256, 0, stream>>>(flux, dtt, fpw, fpb, B0, total);
  gemm(B0, 512, 0, mixw, 512, 1, mixb, B1, 512, 0, 512, 512);
  ln_act_kernel<<<(unsigned)R, 256, 0, stream>>>(B1, 512, B1, 512, mixlnw, mixlnb, 512, 1);
  gemm(B1, 512, 0, few1, 512, 1, feb1, B2, 1024, 0, 1024, 512);
  glu_kernel<<<eb, 256, 0, stream>>>(B2, B0, total);
  ln_act_kernel<<<(unsigned)R, 256, 0, stream>>>(B0, 512, B0, 512, feln1w, feln1b, 512, 0);
  gemm(B0, 512, 0, few2, 512, 1, feb2, B1, 512, 0, 512, 512);
  ln_act_kernel<<<(unsigned)R, 256, 0, stream>>>(B1, 512, B2, 1024, feln2w, feln2b, 512, 1);
  for (int wv = 0; wv < 7; ++wv)
    gemm(B2, 1024, wv - 6, winw + wv, 3584, 7, (wv == 0 ? winb : nullptr),
         B2 + 512, 1024, (wv > 0), 512, 512);
  ln_act_kernel<<<(unsigned)R, 256, 0, stream>>>(B2 + 512, 1024, B2 + 512, 1024,
                                                 winlnw, winlnb, 512, 1);
  // ---- GRU layer 0 ----
  gemm(B2, 1024, 0, g[0][0], 1024, 1, g[0][1], B3, 1536, 0, 1536, 1024);
  gru_scan_kernel<<<48, 128, 0, stream>>>(B3, g[0][2], g[0][3], g[0][4], g[0][5],
                                          g[0][6], g[0][7], g[0][8], g[0][9],
                                          h0, B4, S, stats, sync, TT);
  gemm(B2, 1024, 0, resp, 1024, 1, nullptr, B4, 512, 1, 512, 1024);
  // ---- GRU layer 1 ----
  gemm(B4, 512, 0, g[1][0], 512, 1, g[1][1], B3, 1536, 0, 1536, 512);
  gru_scan_kernel<<<48, 128, 0, stream>>>(B3, g[1][2], g[1][3], g[1][4], g[1][5],
                                          g[1][6], g[1][7], g[1][8], g[1][9],
                                          h1, B5, S, stats + statsL, sync + 8, TT);
  add_kernel<<<eb, 256, 0, stream>>>(B5, B4, total);
  // ---- GRU layer 2 ----
  gemm(B5, 512, 0, g[2][0], 512, 1, g[2][1], B3, 1536, 0, 1536, 512);
  gru_scan_kernel<<<48, 128, 0, stream>>>(B3, g[2][2], g[2][3], g[2][4], g[2][5],
                                          g[2][6], g[2][7], g[2][8], g[2][9],
                                          h2, B0, S, stats + 2 * statsL, sync + 16, TT);
  add_kernel<<<eb, 256, 0, stream>>>(B0, B5, total);
  // ---- attention pooling + classifier ----
  gemm(B0, 512, 0, attw1, 512, 1, attb1, B1, 256, 0, 256, 512);
  tanh_kernel<<<(unsigned)((R * 256 + 255) / 256), 256, 0, stream>>>(B1, (int)(R * 256));
  score_kernel<<<(unsigned)R, 256, 0, stream>>>(B1, attw2, attb2, sbuf);
  softmax_pool_kernel<<<BBATCH, 256, 0, stream>>>(sbuf, B0, pooled, TT);
  cls_kernel<<<1, 128, 0, stream>>>(pooled, clsw, clsb, (float*)d_out);
}
